// Encoder_17626545782821
// MI455X (gfx1250) — matrix operand order repaired and checked by reading, not compiled, not run on hardware
//
#include <hip/hip_runtime.h>
#include <math.h>

// ---------------------------------------------------------------------------
// Types for CDNA5 WMMA (wave32, bf16 inputs, f32 accumulate)
// ---------------------------------------------------------------------------
typedef __bf16 bf16_t;
typedef __attribute__((ext_vector_type(16))) __bf16 v16bf;
typedef __attribute__((ext_vector_type(8)))  __bf16 v8bf;
typedef __attribute__((ext_vector_type(8)))  float  v8f;
typedef __attribute__((ext_vector_type(4)))  float  v4f;
typedef __attribute__((ext_vector_type(4)))  int    v4i;

#define BM 128
#define BN 128
#define BK 32
#define LDSS 40   // LDS row stride in bf16 elems (80B, keeps 16B alignment, skews banks)

// ---------------------------------------------------------------------------
// CDNA5 async global->LDS path (ASYNCcnt).  The ROCm clang builtin takes
// typed int4 pointers: (int4 AS1* gsrc, int4 AS3* ldsdst, imm offset, imm cpol).
// Fall back to inline asm on-device, plain copy for the host-compile parse.
// ---------------------------------------------------------------------------
#if defined(__AMDGCN__) && __has_builtin(__builtin_amdgcn_global_load_async_to_lds_b128) && __has_builtin(__builtin_amdgcn_s_wait_asynccnt)
#define ASYNC_MODE 1
#elif defined(__AMDGCN__)
#define ASYNC_MODE 2
#else
#define ASYNC_MODE 0
#endif

__device__ __forceinline__ void async_copy_b128(const bf16_t* __restrict__ src,
                                                bf16_t* __restrict__ dst) {
#if ASYNC_MODE == 1
    __builtin_amdgcn_global_load_async_to_lds_b128(
        (__attribute__((address_space(1))) v4i*)src,
        (__attribute__((address_space(3))) v4i*)dst, 0, 0);
#elif ASYNC_MODE == 2
    unsigned l = (unsigned)(unsigned long long)dst;     // low 32 bits = LDS addr
    asm volatile("global_load_async_to_lds_b128 %0, %1, off"
                 :: "v"(l), "v"((unsigned long long)src) : "memory");
#else
    *(v8bf*)dst = *(const v8bf*)src;
#endif
}

__device__ __forceinline__ void wait_async_zero() {
#if ASYNC_MODE == 1
    __builtin_amdgcn_s_wait_asynccnt(0);
#elif ASYNC_MODE == 2
    asm volatile("s_wait_asynccnt 0" ::: "memory");
#endif
}

// ---------------------------------------------------------------------------
// Generic GEMM: C[M,N] = act( A[M,K](f32) @ W[K,N] + bias ),  W pre-transposed
// to bf16 Bt[Np][Kp] (Kp = K rounded to 32, Np rounded to 128, zero padded).
// Double-buffered LDS: B tiles arrive via global_load_async_to_lds_b128 while
// WMMA consumes the previous tile; A tiles are f32->bf16 converted in-flight.
// 8 waves/block, each wave a 32x64 tile (2x4 v_wmma_f32_16x16x32_bf16).
// ---------------------------------------------------------------------------
__launch_bounds__(256, 2)
__global__ void k_gemm(const float* __restrict__ A, int lda,
                       const bf16_t* __restrict__ Bt, int ldb /*Kp*/,
                       const float* __restrict__ bias,
                       float* __restrict__ C, int ldc,
                       int M, int N, int K, int relu)
{
    __shared__ bf16_t As[2][BM * LDSS];
    __shared__ bf16_t Bs[2][BN * LDSS];

    const int tid  = threadIdx.x;
    const int lane = tid & 31;
    const int wave = tid >> 5;
    const int wm   = wave & 3;     // wave row   (4 waves in M)
    const int wn   = wave >> 2;    // wave col   (2 waves in N)
    const int hf   = lane >> 4;    // lane half (ISA 16-bit A/B layout)
    const int l16  = lane & 15;

    const int rowBlock = blockIdx.y * BM;
    const int colBlock = blockIdx.x * BN;

    v8f acc[2][4];
    #pragma unroll
    for (int i = 0; i < 2; ++i)
        #pragma unroll
        for (int j = 0; j < 4; ++j)
            #pragma unroll
            for (int r = 0; r < 8; ++r) acc[i][j][r] = 0.0f;

    const int  ksteps = (K + BK - 1) / BK;
    const bool a4     = ((lda & 3) == 0);
    const bool fullM  = (rowBlock + BM <= M);

    // ---- stage A tile (f32 -> bf16), 16 elems/thread ----
    auto stageA = [&](int buf, int kt) {
        const int k0 = kt * BK;
        const int kk = (tid & 7) * 4;   // 0..28
        const int r0 = tid >> 3;        // 0..31
        const bool fast = fullM && a4 && (k0 + BK <= K);
        if (fast) {
            #pragma unroll
            for (int rr = 0; rr < 4; ++rr) {
                const int r = r0 + rr * 32;
                const float* ap = A + (size_t)(rowBlock + r) * lda + k0 + kk;
                v4f v = *(const v4f*)ap;
                bf16_t* d = &As[buf][r * LDSS + kk];
                d[0] = (bf16_t)v.x; d[1] = (bf16_t)v.y;
                d[2] = (bf16_t)v.z; d[3] = (bf16_t)v.w;
            }
        } else {
            #pragma unroll
            for (int rr = 0; rr < 4; ++rr) {
                const int r    = r0 + rr * 32;
                const int grow = rowBlock + r;
                float x0 = 0.f, x1 = 0.f, x2 = 0.f, x3 = 0.f;
                if (grow < M) {
                    const float* ap = A + (size_t)grow * lda + k0 + kk;
                    if (k0 + kk + 0 < K) x0 = ap[0];
                    if (k0 + kk + 1 < K) x1 = ap[1];
                    if (k0 + kk + 2 < K) x2 = ap[2];
                    if (k0 + kk + 3 < K) x3 = ap[3];
                }
                bf16_t* d = &As[buf][r * LDSS + kk];
                d[0] = (bf16_t)x0; d[1] = (bf16_t)x1;
                d[2] = (bf16_t)x2; d[3] = (bf16_t)x3;
            }
        }
    };

    // ---- B tile via async global->LDS (Bt zero-padded to Np mult of 128) ----
    auto issueB = [&](int buf, int kt) {
        const int k0 = kt * BK;
        const int kk = (tid & 3) * 8;   // 0,8,16,24
        const int n0 = tid >> 2;        // 0..63
        #pragma unroll
        for (int nn = 0; nn < 2; ++nn) {
            const int n = n0 + nn * 64;
            const bf16_t* src = Bt + (size_t)(colBlock + n) * ldb + k0 + kk;
            async_copy_b128(src, &Bs[buf][n * LDSS + kk]);
        }
    };

    // prologue: fill buffer 0
    stageA(0, 0);
    issueB(0, 0);
    wait_async_zero();
    __syncthreads();

    for (int kt = 0; kt < ksteps; ++kt) {
        const int cur = kt & 1, nxt = cur ^ 1;

        // overlap: fetch next tiles while computing current
        if (kt + 1 < ksteps) {
            issueB(nxt, kt + 1);
            stageA(nxt, kt + 1);
        }
        if (kt + 2 < ksteps && fullM) {
            __builtin_prefetch(&A[(size_t)(rowBlock + (tid >> 1)) * lda + (size_t)(kt + 2) * BK], 0, 1);
        }

        // ---- fragments per ISA 16-bit layouts ----
        v16bf afrag[2];
        #pragma unroll
        for (int mi = 0; mi < 2; ++mi) {
            const int r = wm * 32 + mi * 16 + l16;
            const bf16_t* ap = &As[cur][r * LDSS];
            v8bf lo = *(const v8bf*)(ap + hf * 8);        // K 0..7  / 8..15
            v8bf hi = *(const v8bf*)(ap + 16 + hf * 8);   // K 16..23 / 24..31
            afrag[mi] = __builtin_shufflevector(lo, hi, 0,1,2,3,4,5,6,7,
                                                        8,9,10,11,12,13,14,15);
        }
        #pragma unroll
        for (int ni = 0; ni < 4; ++ni) {
            const int c = wn * 64 + ni * 16 + l16;
            const bf16_t* bp = &Bs[cur][c * LDSS];
            v8bf lo = *(const v8bf*)(bp + hf * 8);
            v8bf hi = *(const v8bf*)(bp + 16 + hf * 8);
            v16bf bfrag = __builtin_shufflevector(lo, hi, 0,1,2,3,4,5,6,7,
                                                          8,9,10,11,12,13,14,15);
            #pragma unroll
            for (int mi = 0; mi < 2; ++mi) {
                acc[mi][ni] = __builtin_amdgcn_wmma_f32_16x16x32_bf16(
                    false, afrag[mi], false, bfrag, (short)0, acc[mi][ni],
                    false, false);
            }
        }

        if (kt + 1 < ksteps) wait_async_zero();  // next B tile landed in LDS
        __syncthreads();                          // + DScnt for next A tile
    }

    // ---- store C (+bias, optional ReLU).  C layout: lane l -> N=l&15,
    //      VGPR r -> M = (l>>4)*8 + r  (ISA 32-bit C/D layout) ----
    #pragma unroll
    for (int mi = 0; mi < 2; ++mi) {
        const int rbase = rowBlock + wm * 32 + mi * 16 + hf * 8;
        #pragma unroll
        for (int ni = 0; ni < 4; ++ni) {
            const int col = colBlock + wn * 64 + ni * 16 + l16;
            if (col < N) {
                const float bv = bias ? bias[col] : 0.0f;
                #pragma unroll
                for (int r = 0; r < 8; ++r) {
                    const int grow = rbase + r;
                    if (grow < M) {
                        float v = acc[mi][ni][r] + bv;
                        if (relu) v = fmaxf(v, 0.0f);
                        C[(size_t)grow * ldc + col] = v;
                    }
                }
            }
        }
    }
}

// ---------------------------------------------------------------------------
// Small helper kernels
// ---------------------------------------------------------------------------
__global__ void k_fill(float* p, float v, size_t n) {
    size_t i = (size_t)blockIdx.x * blockDim.x + threadIdx.x;
    size_t s = (size_t)gridDim.x * blockDim.x;
    for (; i < n; i += s) p[i] = v;
}

__global__ void k_deg_scatter(const int* __restrict__ col, int E, float* deg) {
    int i = blockIdx.x * blockDim.x + threadIdx.x;
    if (i < E) atomicAdd(&deg[col[i]], 1.0f);
}

__global__ void k_rsqrt_ip(float* p, int n) {
    int i = blockIdx.x * blockDim.x + threadIdx.x;
    if (i < n) p[i] = rsqrtf(p[i]);
}

// W[K,N] f32 -> Wt[Np][Kp] bf16 (transposed + zero padded)
__global__ void k_cvtW(const float* __restrict__ W, int K, int N,
                       bf16_t* __restrict__ Wt, int Kp, int Np) {
    size_t total = (size_t)Np * Kp;
    size_t i = (size_t)blockIdx.x * blockDim.x + threadIdx.x;
    size_t s = (size_t)gridDim.x * blockDim.x;
    for (; i < total; i += s) {
        int n = (int)(i / Kp), k = (int)(i % Kp);
        float v = (k < K && n < N) ? W[(size_t)k * N + n] : 0.0f;
        Wt[i] = (bf16_t)v;
    }
}

// msg = dinv[row]*dinv[col] * hw[row]; scatter-add to out[col]
__global__ void k_edge(const int* __restrict__ row, const int* __restrict__ col,
                       const float* __restrict__ dinv, const float* __restrict__ hw,
                       float* __restrict__ out, int F) {
    int e = blockIdx.x;
    int f = blockIdx.y * 256 + threadIdx.x;
    if (f >= F) return;
    int r = row[e], c = col[e];
    float k = dinv[r] * dinv[c];
    atomicAdd(&out[(size_t)c * F + f], k * hw[(size_t)r * F + f]);
}

// out = relu(out + dinv^2 * hw + bias)
__global__ void k_selfloop(float* __restrict__ out, const float* __restrict__ hw,
                           const float* __restrict__ dinv,
                           const float* __restrict__ bias, size_t n, int F) {
    size_t i = (size_t)blockIdx.x * blockDim.x + threadIdx.x;
    size_t s = (size_t)gridDim.x * blockDim.x;
    for (; i < n; i += s) {
        int nn = (int)(i / F), f = (int)(i % F);
        float d = dinv[nn];
        out[i] = fmaxf(out[i] + d * d * hw[i] + bias[f], 0.0f);
    }
}

__global__ void k_offsets(const int* __restrict__ nn, int B, int* offs) {
    if (blockIdx.x == 0 && threadIdx.x == 0) {
        int a = 0;
        for (int b = 0; b < B; ++b) { offs[b] = a; a += nn[b]; }
        offs[B] = a;
    }
}

__global__ void k_segmax(const float* __restrict__ h, const int* __restrict__ offs,
                         float* __restrict__ x2, int F) {
    int b = blockIdx.x;
    int f = blockIdx.y * 256 + threadIdx.x;
    if (f >= F) return;
    int s = offs[b], e = offs[b + 1];
    float m = -3.4028235e38f;
    for (int i = s; i < e; ++i) m = fmaxf(m, h[(size_t)i * F + f]);
    x2[(size_t)b * F + f] = m;
}

// d_seq base: PAD + seg_enc everywhere
__global__ void k_pad_base(float* __restrict__ out, const float* __restrict__ seg,
                           size_t n, int H) {
    size_t i = (size_t)blockIdx.x * blockDim.x + threadIdx.x;
    size_t s = (size_t)gridDim.x * blockDim.x;
    for (; i < n; i += s) out[i] = -999.0f + seg[(int)(i % H)];
}

// scatter node features: d_seq[pos, batch, :] = h + seg_enc
__global__ void k_scatter_nodes(const float* __restrict__ h, const int* __restrict__ pos,
                                const int* __restrict__ batch, const float* __restrict__ seg,
                                float* __restrict__ out, size_t n, int Bsz, int H) {
    size_t i = (size_t)blockIdx.x * blockDim.x + threadIdx.x;
    size_t s = (size_t)gridDim.x * blockDim.x;
    for (; i < n; i += s) {
        int nn = (int)(i / H), f = (int)(i % H);
        size_t dst = ((size_t)pos[nn] * Bsz + batch[nn]) * H + f;
        out[dst] = h[i] + seg[f];
    }
}

__global__ void k_mask(const int* __restrict__ nn, float* __restrict__ mask, int B, int L) {
    int i = blockIdx.x * blockDim.x + threadIdx.x;
    if (i < B * L) {
        int b = i / L, l = i % L;
        mask[i] = (l >= nn[b]) ? 1.0f : 0.0f;
    }
}

__global__ void k_klreduce(const float* __restrict__ mu, const float* __restrict__ lv,
                           float* __restrict__ acc, size_t n) {
    __shared__ float red[256];
    size_t i = (size_t)blockIdx.x * blockDim.x + threadIdx.x;
    size_t st = (size_t)gridDim.x * blockDim.x;
    float s = 0.f;
    for (; i < n; i += st) {
        float m = mu[i];
        float z = -fabsf(lv[i]);
        s += 1.0f + z - m * m - expf(z);
    }
    red[threadIdx.x] = s;
    __syncthreads();
    for (int o = 128; o > 0; o >>= 1) {
        if ((int)threadIdx.x < o) red[threadIdx.x] += red[threadIdx.x + o];
        __syncthreads();
    }
    if (threadIdx.x == 0) atomicAdd(acc, red[0]);
}

__global__ void k_klfinal(const float* acc, float* out) {
    if (blockIdx.x == 0 && threadIdx.x == 0) out[0] = -0.5f * acc[0] / 64.0f;
}

// amvo = mu + exp(0.5*zlv)*eps + con_emb[b] + affinity[b]; lv buffer overwritten
__global__ void k_amvo(const float* __restrict__ mu, float* __restrict__ lv_io,
                       const float* __restrict__ eps, const float* __restrict__ cemb,
                       const float* __restrict__ aff, size_t n, int Bsz, int H) {
    size_t i = (size_t)blockIdx.x * blockDim.x + threadIdx.x;
    size_t st = (size_t)gridDim.x * blockDim.x;
    for (; i < n; i += st) {
        int f = (int)(i % H);
        int b = (int)((i / H) % Bsz);
        float z = -fabsf(lv_io[i]);
        float v = mu[i] + expf(0.5f * z) * eps[i];
        lv_io[i] = v + cemb[(size_t)b * H + f] + aff[b];
    }
}

// ---------------------------------------------------------------------------
// Host orchestration
// ---------------------------------------------------------------------------
static inline size_t alignUp(size_t x, size_t a) { return (x + a - 1) & ~(a - 1); }
static inline unsigned gblk(size_t n) {
    size_t b = (n + 255) / 256;
    return (unsigned)(b > 65536 ? 65536 : b);
}

extern "C" void kernel_launch(void* const* d_in, const int* in_sizes, int n_in,
                              void* d_out, int out_size, void* d_ws, size_t ws_size,
                              hipStream_t stream) {
    (void)n_in; (void)out_size; (void)ws_size;
    const int DF = 94, H = 376, L = 64, B = 2048, FINAL = 256;
    const int N = in_sizes[0] / DF;
    const int E = in_sizes[4] / 2;
    const size_t LBH = (size_t)L * B * H;

    // inputs
    const float* x      = (const float*)d_in[0];
    const float* con    = (const float*)d_in[1];
    const float* aff    = (const float*)d_in[2];
    const float* eps    = (const float*)d_in[3];
    const int*   ei     = (const int*)d_in[4];
    const int*   batch  = (const int*)d_in[5];
    const int*   nnodes = (const int*)d_in[6];
    const int*   pos    = (const int*)d_in[7];
    const float* seg    = (const float*)d_in[8];
    const float *W1 = (const float*)d_in[9],  *b1 = (const float*)d_in[10];
    const float *W2 = (const float*)d_in[11], *b2 = (const float*)d_in[12];
    const float *W3 = (const float*)d_in[13], *b3 = (const float*)d_in[14];
    const float *cW = (const float*)d_in[15], *cb = (const float*)d_in[16];
    const float *mW1 = (const float*)d_in[17], *mb1 = (const float*)d_in[18];
    const float *mW2 = (const float*)d_in[19], *mb2 = (const float*)d_in[20];
    const float *vW1 = (const float*)d_in[21], *vb1 = (const float*)d_in[22];
    const float *vW2 = (const float*)d_in[23], *vb2 = (const float*)d_in[24];
    const float *f1W = (const float*)d_in[25], *f1b = (const float*)d_in[26];
    const float *f2W = (const float*)d_in[27], *f2b = (const float*)d_in[28];
    const int* erow = ei;
    const int* ecol = ei + E;

    // outputs (return order: d_seq, amvo, mask, pmvo, kl)
    float* out_dseq = (float*)d_out;
    float* out_amvo = out_dseq + LBH;
    float* out_mask = out_amvo + LBH;
    float* out_pmvo = out_mask + (size_t)B * L;
    float* out_kl   = out_pmvo + (size_t)B * FINAL;

    // workspace bump allocator
    char* wsb = (char*)d_ws;
    size_t off = 0;
    auto alloc = [&](size_t bytes) -> void* {
        off = alignUp(off, 256);
        void* p = wsb + off;
        off += bytes;
        return p;
    };

    // Big union region: GCN node buffers (3 x N*H) time-aliased with
    // the two MLP intermediates (2 x L*B*H).  2*LBH > 3*N*H holds here.
    const size_t Nh = (size_t)N * H;
    float* region = (float*)alloc(2 * LBH * sizeof(float));
    float* bufP = region;            // node buf 0
    float* bufW = region + Nh;       // node buf 1 (h @ W scratch)
    float* bufO = region + 2 * Nh;   // node buf 2
    float* big_t  = region;          // MLP intermediate (after GCN dead)
    float* big_mu = region + LBH;    // mu (after h3 dead)

    float* dinv  = (float*)alloc((size_t)N * 4);
    float* x2    = (float*)alloc((size_t)B * H * 4);
    float* t_fc  = (float*)alloc((size_t)B * 1024 * 4);
    float* cemb  = (float*)alloc((size_t)B * H * 4);
    int*   offs  = (int*)alloc((size_t)(B + 1) * 4);
    float* klsum = (float*)alloc(4);

    // Weights -> transposed bf16, K padded to 32, N padded to 128 (zero fill)
    // so async B-tile loads never run out of bounds.
    auto rupK = [](int v) { return (v + 31) / 32 * 32; };
    auto rupN = [](int v) { return (v + 127) / 128 * 128; };
    struct WDesc { const float* W; int K, N, Kp, Np; bf16_t* t; };
    WDesc wd[10] = {
        { W1,  DF,     2*DF, rupK(DF),     rupN(2*DF), nullptr },
        { W2,  2*DF,   3*DF, rupK(2*DF),   rupN(3*DF), nullptr },
        { W3,  3*DF,   4*DF, rupK(3*DF),   rupN(4*DF), nullptr },
        { cW,  96*107, H,    rupK(96*107), rupN(H),    nullptr },
        { mW1, H, H, rupK(H), rupN(H), nullptr },
        { mW2, H, H, rupK(H), rupN(H), nullptr },
        { vW1, H, H, rupK(H), rupN(H), nullptr },
        { vW2, H, H, rupK(H), rupN(H), nullptr },
        { f1W, H,    1024, rupK(H),    1024, nullptr },
        { f2W, 1024, 256,  rupK(1024), 256,  nullptr },
    };
    for (int i = 0; i < 10; ++i)
        wd[i].t = (bf16_t*)alloc((size_t)wd[i].Kp * wd[i].Np * sizeof(bf16_t));

    auto gemm = [&](const float* A, int lda, const bf16_t* Bt, int Kp,
                    const float* bias, float* C, int ldc,
                    int M, int Nn, int K, int relu) {
        dim3 g((Nn + BN - 1) / BN, (M + BM - 1) / BM);
        k_gemm<<<g, 256, 0, stream>>>(A, lda, Bt, Kp, bias, C, ldc, M, Nn, K, relu);
    };

    // ---- degrees -> dinv ----
    k_fill<<<gblk(N), 256, 0, stream>>>(dinv, 1.0f, (size_t)N);
    k_deg_scatter<<<(E + 255) / 256, 256, 0, stream>>>(ecol, E, dinv);
    k_rsqrt_ip<<<(N + 255) / 256, 256, 0, stream>>>(dinv, N);

    // ---- weight conversion (f32 -> transposed, padded bf16) ----
    for (int i = 0; i < 10; ++i) {
        size_t t = (size_t)wd[i].Kp * wd[i].Np;
        k_cvtW<<<gblk(t), 256, 0, stream>>>(wd[i].W, wd[i].K, wd[i].N,
                                            wd[i].t, wd[i].Kp, wd[i].Np);
    }

    // ---- GCN layer 1: x[N,94] -> h1[N,188] (in bufO) ----
    gemm(x, DF, wd[0].t, wd[0].Kp, nullptr, bufW, 2*DF, N, 2*DF, DF, 0);
    k_fill<<<gblk((size_t)N * 2*DF), 256, 0, stream>>>(bufO, 0.0f, (size_t)N * 2*DF);
    k_edge<<<dim3(E, 1), 256, 0, stream>>>(erow, ecol, dinv, bufW, bufO, 2*DF);
    k_selfloop<<<gblk((size_t)N * 2*DF), 256, 0, stream>>>(bufO, bufW, dinv, b1,
                                                           (size_t)N * 2*DF, 2*DF);
    // ---- GCN layer 2: h1 -> h2[N,282] (in bufP) ----
    gemm(bufO, 2*DF, wd[1].t, wd[1].Kp, nullptr, bufW, 3*DF, N, 3*DF, 2*DF, 0);
    k_fill<<<gblk((size_t)N * 3*DF), 256, 0, stream>>>(bufP, 0.0f, (size_t)N * 3*DF);
    k_edge<<<dim3(E, 2), 256, 0, stream>>>(erow, ecol, dinv, bufW, bufP, 3*DF);
    k_selfloop<<<gblk((size_t)N * 3*DF), 256, 0, stream>>>(bufP, bufW, dinv, b2,
                                                           (size_t)N * 3*DF, 3*DF);
    // ---- GCN layer 3: h2 -> h3[N,376] (in bufO) ----
    gemm(bufP, 3*DF, wd[2].t, wd[2].Kp, nullptr, bufW, H, N, H, 3*DF, 0);
    k_fill<<<gblk(Nh), 256, 0, stream>>>(bufO, 0.0f, Nh);
    k_edge<<<dim3(E, 2), 256, 0, stream>>>(erow, ecol, dinv, bufW, bufO, H);
    k_selfloop<<<gblk(Nh), 256, 0, stream>>>(bufO, bufW, dinv, b3, Nh, H);

    // ---- global max pool + FC head (pmvo) ----
    k_offsets<<<1, 1, 0, stream>>>(nnodes, B, offs);
    k_segmax<<<dim3(B, 2), 256, 0, stream>>>(bufO, offs, x2, H);
    gemm(x2, H, wd[8].t, wd[8].Kp, f1b, t_fc, 1024, B, 1024, H, 1);
    gemm(t_fc, 1024, wd[9].t, wd[9].Kp, f2b, out_pmvo, FINAL, B, FINAL, 1024, 0);

    // ---- pad + seg_enc -> d_seq; mask ----
    k_pad_base<<<gblk(LBH), 256, 0, stream>>>(out_dseq, seg, LBH, H);
    k_scatter_nodes<<<gblk(Nh), 256, 0, stream>>>(bufO, pos, batch, seg,
                                                  out_dseq, Nh, B, H);
    k_mask<<<(B * L + 255) / 256, 256, 0, stream>>>(nnodes, out_mask, B, L);

    // ---- conditioning embedding ----
    gemm(con, 96 * 107, wd[3].t, wd[3].Kp, cb, cemb, H, B, H, 96 * 107, 0);

    // ---- mu / logvar MLPs (dominant GEMMs; bufO/h3 is dead past this point) ----
    const int M = L * B;  // 131072
    gemm(out_dseq, H, wd[4].t, wd[4].Kp, mb1, big_t, H, M, H, H, 1);
    gemm(big_t,   H, wd[5].t, wd[5].Kp, mb2, big_mu, H, M, H, H, 0);
    gemm(out_dseq, H, wd[6].t, wd[6].Kp, vb1, big_t, H, M, H, H, 1);
    gemm(big_t,   H, wd[7].t, wd[7].Kp, vb2, out_amvo /*logvar staging*/, H, M, H, H, 0);

    // ---- KL, then reparameterize + broadcast into amvo (in place) ----
    k_fill<<<1, 1, 0, stream>>>(klsum, 0.0f, 1);
    k_klreduce<<<4096, 256, 0, stream>>>(big_mu, out_amvo, klsum, LBH);
    k_klfinal<<<1, 1, 0, stream>>>(klsum, out_kl);
    k_amvo<<<gblk(LBH), 256, 0, stream>>>(big_mu, out_amvo, eps, cemb, aff, LBH, B, H);
}
